// dba_attention_14267881357689
// MI455X (gfx1250) — compile-verified, matrix operand order checked by reading, not run on hardware
//
#include <hip/hip_runtime.h>
#include <hip/hip_bf16.h>

// ---------------- types / helpers ----------------
typedef __attribute__((ext_vector_type(16))) __bf16 v16bf;
typedef __attribute__((ext_vector_type(8)))  float  v8f;

#define SCALE_F 0.5773502691896258f   // 3^-0.5
#define NEGBIG  -3.0e38f

struct FragBits { uint4 lo; uint4 hi; };
union U8 { uint4 u; __bf16 h[8]; };

__device__ __forceinline__ v16bf load_frag(const __bf16* p0, const __bf16* p1) {
  FragBits f;
  f.lo = *(const uint4*)p0;
  f.hi = *(const uint4*)p1;
  return __builtin_bit_cast(v16bf, f);
}

__device__ __forceinline__ v8f wmma_bf16(v16bf a, v16bf b, v8f c) {
  // D(f32 16x16) = A(bf16 16x32) * B(bf16 32x16) + C
  return __builtin_amdgcn_wmma_f32_16x16x32_bf16(false, a, false, b, (short)0, c,
                                                 false, false);
}

// ---------------- f32 -> bf16 conversion ----------------
__global__ void __launch_bounds__(256)
k_cvt_bf16(const float* __restrict__ in, __bf16* __restrict__ out, int n4)
{
  int i = blockIdx.x * 256 + threadIdx.x;
  if (i >= n4) return;
  float4 v = ((const float4*)in)[i];
  out[i * 4 + 0] = (__bf16)v.x;
  out[i * 4 + 1] = (__bf16)v.y;
  out[i * 4 + 2] = (__bf16)v.z;
  out[i * 4 + 3] = (__bf16)v.w;
}

// ---------------- generic WMMA GEMM: C[M,N] = A[M,K] @ B[N,K]^T + bias ----------------
// Register-blocked: each wave computes a 64x64 tile (16 WMMAs / K-step,
// 8 A-loads + 8 B-loads / K-step => 1 b128 load per WMMA).
// grid = (N/64, M/512), block = 256 (8 waves, consecutive 64-row tiles, same cols)
__global__ void __launch_bounds__(256)
k_gemm_bf16(const __bf16* __restrict__ A, const __bf16* __restrict__ B,
            const float* __restrict__ bias, float* __restrict__ Cf,
            __bf16* __restrict__ Cb, int N, int K)
{
  const int lane  = threadIdx.x & 31;
  const int wave  = threadIdx.x >> 5;
  const int mrow  = lane & 15;
  const int kb    = (lane >> 4) << 3;      // 0 or 8 (K-half pattern)
  const int mbase = (blockIdx.y * 8 + wave) * 64;
  const int nbase = blockIdx.x * 64;

  const __bf16* ap = A + (size_t)(mbase + mrow) * K + kb;
  const __bf16* bp = B + (size_t)(nbase + mrow) * K + kb;
  const size_t rs16 = (size_t)16 * K;      // 16-row stride in elements

  v8f c[4][4];
  #pragma unroll
  for (int im = 0; im < 4; ++im)
    #pragma unroll
    for (int in = 0; in < 4; ++in)
      c[im][in] = v8f{0.f,0.f,0.f,0.f,0.f,0.f,0.f,0.f};

  for (int k0 = 0; k0 < K; k0 += 32) {
    v16bf a[4], b[4];
    #pragma unroll
    for (int i = 0; i < 4; ++i) {
      a[i] = load_frag(ap + i * rs16 + k0, ap + i * rs16 + k0 + 16);
      b[i] = load_frag(bp + i * rs16 + k0, bp + i * rs16 + k0 + 16);
    }
    #pragma unroll
    for (int im = 0; im < 4; ++im)
      #pragma unroll
      for (int in = 0; in < 4; ++in)
        c[im][in] = wmma_bf16(a[im], b[in], c[im][in]);
  }

  float bn[4];
  #pragma unroll
  for (int in = 0; in < 4; ++in) bn[in] = bias[nbase + in * 16 + mrow];

  #pragma unroll
  for (int im = 0; im < 4; ++im)
    #pragma unroll
    for (int in = 0; in < 4; ++in) {
      const int col = nbase + in * 16 + mrow;
      #pragma unroll
      for (int i = 0; i < 8; ++i) {
        float v = c[im][in][i] + bn[in];
        size_t idx = (size_t)(mbase + im * 16 + kb + i) * N + col;
        if (Cf) Cf[idx] = v;
        else    Cb[idx] = (__bf16)v;
      }
    }
}

// ---------------- small projection: Y[M,24] = (A[M,512] @ W[24,512]^T + b) * scale ----------------
__global__ void __launch_bounds__(256)
k_proj24(const float* __restrict__ A, const float* __restrict__ W,
         const float* __restrict__ bias, float scale, float* __restrict__ Y, int M)
{
  int idx = blockIdx.x * 256 + threadIdx.x;
  if (idx >= M * 24) return;
  int m = idx / 24, j = idx - m * 24;
  const float4* a  = (const float4*)(A + (size_t)m * 512);
  const float4* wv = (const float4*)(W + (size_t)j * 512);
  float s = 0.f;
  #pragma unroll 4
  for (int k = 0; k < 128; ++k) {
    float4 av = a[k], wt = wv[k];
    s = fmaf(av.x, wt.x, s); s = fmaf(av.y, wt.y, s);
    s = fmaf(av.z, wt.z, s); s = fmaf(av.w, wt.w, s);
  }
  Y[idx] = (s + bias[j]) * scale;
}

// ---------------- stage 1: flash attention L(2048) -> P(128), per bh ----------------
// grid = 128 (bh), block = 256 (wave w owns p rows [16w,16w+16))
__global__ void __launch_bounds__(256)
k_stage1(const float* __restrict__ pkf, const float* __restrict__ pqf,
         const __bf16* __restrict__ pvb, const unsigned char* __restrict__ mask,
         float* __restrict__ pc_out, __bf16* __restrict__ pc_b16)
{
  const int bh = blockIdx.x;
  const int b = bh >> 3, h = bh & 7;
  const int wave = threadIdx.x >> 5, lane = threadIdx.x & 31;
  const int mrow = lane & 15;
  const int kb   = (lane >> 4) << 3;
  const int prow = wave * 16 + mrow;

  __shared__ float  s_pk[32 * 3];
  __shared__ float  s_mb[32];
  __shared__ __bf16 s_pvT[64][32];   // pv chunk transposed: [v][l]

  const float q0 = pqf[((size_t)prow * 16 + b) * 24 + h * 3 + 0];
  const float q1 = pqf[((size_t)prow * 16 + b) * 24 + h * 3 + 1];
  const float q2 = pqf[((size_t)prow * 16 + b) * 24 + h * 3 + 2];

  v8f c[4];
  #pragma unroll
  for (int v = 0; v < 4; ++v) c[v] = v8f{0.f,0.f,0.f,0.f,0.f,0.f,0.f,0.f};
  float m_run = NEGBIG, s_run = 0.f;

  for (int l0 = 0; l0 < 2048; l0 += 32) {
    __syncthreads();
    const int t = threadIdx.x;
    { // stage pv chunk, transposing [l][v] -> [v][l]
      const int l = t >> 3, vg = (t & 7) << 3;
      U8 u;
      u.u = *(const uint4*)(pvb + ((size_t)(l0 + l) * 16 + b) * 512 + h * 64 + vg);
      #pragma unroll
      for (int i = 0; i < 8; ++i) s_pvT[vg + i][l] = u.h[i];
    }
    if (t < 96) {
      s_pk[t] = pkf[((size_t)(l0 + t / 3) * 16 + b) * 24 + h * 3 + (t % 3)];
    } else if (t < 128) {
      int l = t - 96;
      s_mb[l] = mask[b * 2048 + l0 + l] ? NEGBIG : 0.f;
    }
    __syncthreads();

    // scores: lane owns row p=mrow, l-offsets {kb..kb+7, kb+16..kb+23} (A-fragment layout)
    float sc[16];
    float mloc = NEGBIG;
    #pragma unroll
    for (int t8 = 0; t8 < 8; ++t8) {
      int la = kb + t8, lb2 = kb + 16 + t8;
      float sa = fmaf(q0, s_pk[la * 3], fmaf(q1, s_pk[la * 3 + 1], q2 * s_pk[la * 3 + 2])) + s_mb[la];
      float sb = fmaf(q0, s_pk[lb2 * 3], fmaf(q1, s_pk[lb2 * 3 + 1], q2 * s_pk[lb2 * 3 + 2])) + s_mb[lb2];
      sc[t8] = sa; sc[8 + t8] = sb;
      mloc = fmaxf(mloc, fmaxf(sa, sb));
    }
    mloc = fmaxf(mloc, __shfl_xor(mloc, 16));
    const float mnew = fmaxf(m_run, mloc);
    const float corr = __expf(m_run - mnew);

    float ssum = 0.f;
    v16bf a;
    #pragma unroll
    for (int i = 0; i < 16; ++i) {
      float p = __expf(sc[i] - mnew);
      ssum += p;
      a[i] = (__bf16)p;
    }
    ssum += __shfl_xor(ssum, 16);
    s_run = s_run * corr + ssum;
    m_run = mnew;

    // broadcast per-row rescale into C-fragment rows (row = kb+i, held by lane kb+i)
    float corr_i[8];
    #pragma unroll
    for (int i = 0; i < 8; ++i) corr_i[i] = __shfl(corr, kb + i);
    #pragma unroll
    for (int v = 0; v < 4; ++v)
      #pragma unroll
      for (int i = 0; i < 8; ++i) c[v][i] *= corr_i[i];

    #pragma unroll
    for (int v = 0; v < 4; ++v) {
      v16bf bf = load_frag(&s_pvT[v * 16 + mrow][kb], &s_pvT[v * 16 + mrow][kb + 16]);
      c[v] = wmma_bf16(a, bf, c[v]);
    }
  }

  float inv_i[8];
  #pragma unroll
  for (int i = 0; i < 8; ++i) inv_i[i] = 1.f / __shfl(s_run, kb + i);
  #pragma unroll
  for (int v = 0; v < 4; ++v) {
    #pragma unroll
    for (int i = 0; i < 8; ++i) {
      int p = wave * 16 + kb + i;
      size_t idx = ((size_t)p * 16 + b) * 512 + h * 64 + v * 16 + mrow;
      float val = c[v][i] * inv_i[i];
      pc_out[idx] = val;
      pc_b16[idx] = (__bf16)val;
    }
  }
}

// ---------------- stage 2: attention within P, per bh ----------------
__global__ void __launch_bounds__(256)
k_stage2(const float* __restrict__ q2p, const float* __restrict__ k2p,
         const __bf16* __restrict__ v2b, float* __restrict__ attn)
{
  const int bh = blockIdx.x;
  const int b = bh >> 3, h = bh & 7;
  const int wave = threadIdx.x >> 5, lane = threadIdx.x & 31;
  const int mrow = lane & 15, kb = (lane >> 4) << 3;

  __shared__ float  s_k[128 * 3];
  __shared__ __bf16 s_vT[64][128];

  const int t = threadIdx.x;
  if (t < 128) {
    #pragma unroll
    for (int j = 0; j < 3; ++j)
      s_k[t * 3 + j] = k2p[((size_t)t * 16 + b) * 24 + h * 3 + j];
  }
  for (int pi = t >> 3; pi < 128; pi += 32) {
    const int vg = (t & 7) << 3;
    U8 u;
    u.u = *(const uint4*)(v2b + ((size_t)pi * 16 + b) * 512 + h * 64 + vg);
    #pragma unroll
    for (int i = 0; i < 8; ++i) s_vT[vg + i][pi] = u.h[i];
  }
  __syncthreads();

  const int prow = wave * 16 + mrow;
  const float q0 = q2p[((size_t)prow * 16 + b) * 24 + h * 3 + 0];
  const float q1 = q2p[((size_t)prow * 16 + b) * 24 + h * 3 + 1];
  const float q2v = q2p[((size_t)prow * 16 + b) * 24 + h * 3 + 2];

  float sc[64];
  float mloc = NEGBIG;
  #pragma unroll
  for (int kk = 0; kk < 4; ++kk)
    #pragma unroll
    for (int t8 = 0; t8 < 8; ++t8) {
      int la  = kk * 32 + kb + t8;
      int lb2 = la + 16;
      float sa = fmaf(q0, s_k[la * 3],  fmaf(q1, s_k[la * 3 + 1],  q2v * s_k[la * 3 + 2]));
      float sb = fmaf(q0, s_k[lb2 * 3], fmaf(q1, s_k[lb2 * 3 + 1], q2v * s_k[lb2 * 3 + 2]));
      sc[kk * 16 + t8] = sa; sc[kk * 16 + 8 + t8] = sb;
      mloc = fmaxf(mloc, fmaxf(sa, sb));
    }
  mloc = fmaxf(mloc, __shfl_xor(mloc, 16));
  float ssum = 0.f;
  #pragma unroll
  for (int i = 0; i < 64; ++i) { sc[i] = __expf(sc[i] - mloc); ssum += sc[i]; }
  ssum += __shfl_xor(ssum, 16);
  const float inv = 1.f / ssum;

  v8f c[4];
  #pragma unroll
  for (int v = 0; v < 4; ++v) c[v] = v8f{0.f,0.f,0.f,0.f,0.f,0.f,0.f,0.f};
  #pragma unroll
  for (int kk = 0; kk < 4; ++kk) {
    v16bf a;
    #pragma unroll
    for (int i = 0; i < 16; ++i) a[i] = (__bf16)(sc[kk * 16 + i] * inv);
    #pragma unroll
    for (int vt = 0; vt < 4; ++vt) {
      v16bf bf = load_frag(&s_vT[vt * 16 + mrow][kk * 32 + kb],
                           &s_vT[vt * 16 + mrow][kk * 32 + kb + 16]);
      c[vt] = wmma_bf16(a, bf, c[vt]);
    }
  }
  #pragma unroll
  for (int vt = 0; vt < 4; ++vt)
    #pragma unroll
    for (int i = 0; i < 8; ++i) {
      int p = wave * 16 + kb + i;
      attn[((size_t)p * 16 + b) * 512 + h * 64 + vt * 16 + mrow] = c[vt][i];
    }
}

// ---------------- layernorm over D=512, output bf16 ----------------
__global__ void __launch_bounds__(256)
k_layernorm(const float* __restrict__ X, const float* __restrict__ g,
            const float* __restrict__ bt, __bf16* __restrict__ Y)
{
  const int row = blockIdx.x;
  const int tid = threadIdx.x;
  const float* x = X + (size_t)row * 512;
  float a = x[tid], b = x[tid + 256];
  __shared__ float red[256];
  red[tid] = a + b;
  __syncthreads();
  for (int s = 128; s > 0; s >>= 1) { if (tid < s) red[tid] += red[tid + s]; __syncthreads(); }
  const float mean = red[0] * (1.f / 512.f);
  __syncthreads();
  float d0 = a - mean, d1 = b - mean;
  red[tid] = d0 * d0 + d1 * d1;
  __syncthreads();
  for (int s = 128; s > 0; s >>= 1) { if (tid < s) red[tid] += red[tid + s]; __syncthreads(); }
  const float rstd = rsqrtf(red[0] * (1.f / 512.f) + 1e-5f);
  Y[(size_t)row * 512 + tid]       = (__bf16)(d0 * rstd * g[tid] + bt[tid]);
  Y[(size_t)row * 512 + tid + 256] = (__bf16)(d1 * rstd * g[tid + 256] + bt[tid + 256]);
}

// ---------------- stage 3: softmax(q_len) @ attn_ln, per (bh, l-tiles) ----------------
// grid = (16, 128): block owns 8 l-tiles of one bh; attn_ln^T staged once in LDS
__global__ void __launch_bounds__(256)
k_stage3(const float* __restrict__ qlen, const __bf16* __restrict__ aln,
         __bf16* __restrict__ Yb)
{
  const int bh = blockIdx.y, b = bh >> 3, h = bh & 7;
  const int wave = threadIdx.x >> 5, lane = threadIdx.x & 31;
  const int mrow = lane & 15, kb = (lane >> 4) << 3;

  __shared__ __bf16 s_aT[64][128];
  const int t = threadIdx.x;
  for (int pi = t >> 3; pi < 128; pi += 32) {
    const int vg = (t & 7) << 3;
    U8 u;
    u.u = *(const uint4*)(aln + ((size_t)pi * 16 + b) * 512 + h * 64 + vg);
    #pragma unroll
    for (int i = 0; i < 8; ++i) s_aT[vg + i][pi] = u.h[i];
  }
  __syncthreads();

  const int lrow = (blockIdx.x * 8 + wave) * 16 + mrow;
  const float* qr = qlen + ((size_t)lrow * 16 + b) * 1024 + h * 128;

  float sc[64];
  #pragma unroll
  for (int kk = 0; kk < 4; ++kk) {
    float4 u0 = *(const float4*)(qr + kk * 32 + kb);
    float4 u1 = *(const float4*)(qr + kk * 32 + kb + 4);
    float4 u2 = *(const float4*)(qr + kk * 32 + kb + 16);
    float4 u3 = *(const float4*)(qr + kk * 32 + kb + 20);
    sc[kk*16+0]=u0.x; sc[kk*16+1]=u0.y; sc[kk*16+2]=u0.z;  sc[kk*16+3]=u0.w;
    sc[kk*16+4]=u1.x; sc[kk*16+5]=u1.y; sc[kk*16+6]=u1.z;  sc[kk*16+7]=u1.w;
    sc[kk*16+8]=u2.x; sc[kk*16+9]=u2.y; sc[kk*16+10]=u2.z; sc[kk*16+11]=u2.w;
    sc[kk*16+12]=u3.x; sc[kk*16+13]=u3.y; sc[kk*16+14]=u3.z; sc[kk*16+15]=u3.w;
  }
  float mloc = NEGBIG;
  #pragma unroll
  for (int i = 0; i < 64; ++i) mloc = fmaxf(mloc, sc[i]);
  mloc = fmaxf(mloc, __shfl_xor(mloc, 16));
  float ssum = 0.f;
  #pragma unroll
  for (int i = 0; i < 64; ++i) { sc[i] = __expf(sc[i] - mloc); ssum += sc[i]; }
  ssum += __shfl_xor(ssum, 16);
  const float inv = 1.f / ssum;

  v8f c[4];
  #pragma unroll
  for (int v = 0; v < 4; ++v) c[v] = v8f{0.f,0.f,0.f,0.f,0.f,0.f,0.f,0.f};
  #pragma unroll
  for (int kk = 0; kk < 4; ++kk) {
    v16bf a;
    #pragma unroll
    for (int i = 0; i < 16; ++i) a[i] = (__bf16)(sc[kk * 16 + i] * inv);
    #pragma unroll
    for (int vt = 0; vt < 4; ++vt) {
      v16bf bf = load_frag(&s_aT[vt * 16 + mrow][kk * 32 + kb],
                           &s_aT[vt * 16 + mrow][kk * 32 + kb + 16]);
      c[vt] = wmma_bf16(a, bf, c[vt]);
    }
  }
  #pragma unroll
  for (int vt = 0; vt < 4; ++vt)
    #pragma unroll
    for (int i = 0; i < 8; ++i) {
      int l = (blockIdx.x * 8 + wave) * 16 + kb + i;
      Yb[((size_t)l * 16 + b) * 512 + h * 64 + vt * 16 + mrow] = (__bf16)c[vt][i];
    }
}

// ---------------- host launcher ----------------
extern "C" void kernel_launch(void* const* d_in, const int* in_sizes, int n_in,
                              void* d_out, int out_size, void* d_ws, size_t ws_size,
                              hipStream_t stream)
{
  (void)in_sizes; (void)n_in; (void)out_size; (void)ws_size;

  const float* x   = (const float*)d_in[0];
  const float* px  = (const float*)d_in[1];
  const unsigned char* mask = (const unsigned char*)d_in[2];
  const float* Wpq = (const float*)d_in[3];
  const float* bpq = (const float*)d_in[4];
  const float* Wq  = (const float*)d_in[5];
  const float* bq  = (const float*)d_in[6];
  const float* Wql = (const float*)d_in[7];
  const float* bql = (const float*)d_in[8];
  const float* Wpk = (const float*)d_in[9];
  const float* bpk = (const float*)d_in[10];
  const float* Wpv = (const float*)d_in[11];
  const float* bpv = (const float*)d_in[12];
  const float* Wk  = (const float*)d_in[13];
  const float* bk  = (const float*)d_in[14];
  const float* Wv  = (const float*)d_in[15];
  const float* bv  = (const float*)d_in[16];
  const float* Wo  = (const float*)d_in[17];
  const float* bo  = (const float*)d_in[18];
  const float* gamma = (const float*)d_in[19];
  const float* beta  = (const float*)d_in[20];

  float* out = (float*)d_out;                      // [L*B*D] = 16,777,216 f32
  float* pc  = out + (size_t)2048 * 16 * 512;      // [P*B*D] =  1,048,576 f32

  char* w = (char*)d_ws;
  auto take = [&](size_t bytes) {
    char* p = w;
    w += (bytes + 255) & ~(size_t)255;
    return p;
  };
  __bf16* xb      = (__bf16*)take((size_t)16777216 * 2);   // x bf16
  __bf16* wpvb    = (__bf16*)take((size_t)262144 * 2);
  __bf16* wqlb    = (__bf16*)take((size_t)524288 * 2);
  __bf16* wvb     = (__bf16*)take((size_t)262144 * 2);
  __bf16* wob     = (__bf16*)take((size_t)262144 * 2);
  __bf16* pvb     = (__bf16*)take((size_t)16777216 * 2);   // pv bf16 [L*B,512]
  float*  qlen    = (float*) take((size_t)33554432 * 4);   // q_len f32 [L*B,1024]
  float*  pkf     = (float*) take((size_t)786432 * 4);     // pk [L*B,24]
  float*  pqf     = (float*) take((size_t)49152 * 4);      // pq [P*B,24] (scaled)
  __bf16* pcb     = (__bf16*)take((size_t)1048576 * 2);    // pc bf16
  float*  k2      = (float*) take((size_t)49152 * 4);
  float*  q2      = (float*) take((size_t)49152 * 4);      // (scaled)
  __bf16* v2b     = (__bf16*)take((size_t)1048576 * 2);
  float*  attnf   = (float*) take((size_t)1048576 * 4);
  __bf16* attnlnb = (__bf16*)take((size_t)1048576 * 2);
  __bf16* yb      = (__bf16*)take((size_t)16777216 * 2);

  // 1) bf16 conversions
  k_cvt_bf16<<<16384, 256, 0, stream>>>(x,   xb,   4194304);
  k_cvt_bf16<<<256,   256, 0, stream>>>(Wpv, wpvb, 65536);
  k_cvt_bf16<<<512,   256, 0, stream>>>(Wql, wqlb, 131072);
  k_cvt_bf16<<<256,   256, 0, stream>>>(Wv,  wvb,  65536);
  k_cvt_bf16<<<256,   256, 0, stream>>>(Wo,  wob,  65536);

  // 2) big GEMMs from x   (grid = (N/64, M/512))
  k_gemm_bf16<<<dim3(8, 64), 256, 0, stream>>>(xb, wpvb, bpv, nullptr, pvb, 512, 512);
  k_gemm_bf16<<<dim3(16, 64), 256, 0, stream>>>(xb, wqlb, bql, qlen, nullptr, 1024, 512);

  // 3) small projections pk, pq
  k_proj24<<<3072, 256, 0, stream>>>(x,  Wpk, bpk, 1.0f,    pkf, 32768);
  k_proj24<<<192,  256, 0, stream>>>(px, Wpq, bpq, SCALE_F, pqf, 2048);

  // 4) stage-1 flash attention -> pc (output #2) + pc bf16
  k_stage1<<<128, 256, 0, stream>>>(pkf, pqf, pvb, mask, pc, pcb);

  // 5) stage-2 projections + attention + layernorm
  k_proj24<<<192, 256, 0, stream>>>(pc, Wk, bk, 1.0f,    k2, 2048);
  k_proj24<<<192, 256, 0, stream>>>(pc, Wq, bq, SCALE_F, q2, 2048);
  k_gemm_bf16<<<dim3(8, 4), 256, 0, stream>>>(pcb, wvb, bv, nullptr, v2b, 512, 512);
  k_stage2<<<128, 256, 0, stream>>>(q2, k2, v2b, attnf);
  k_layernorm<<<2048, 256, 0, stream>>>(attnf, gamma, beta, attnlnb);

  // 6) stage-3 re-expansion P -> L
  k_stage3<<<dim3(16, 128), 256, 0, stream>>>(qlen, attnlnb, yb);

  // 7) output projection
  k_gemm_bf16<<<dim3(8, 64), 256, 0, stream>>>(yb, wob, bo, out, nullptr, 512, 512);
}
